// SelectiveWKV_1B_2018634629633
// MI455X (gfx1250) — matrix operand order repaired and checked by reading, not compiled, not run on hardware
//
#include <hip/hip_runtime.h>

typedef __attribute__((ext_vector_type(16))) __bf16 v16bf;
typedef __attribute__((ext_vector_type(8)))  float  v8f;

#define B_SZ   2
#define T_SZ   2048
#define C_SZ   2048
#define H_SZ   32
#define HS_SZ  64
#define M_ROWS (B_SZ * T_SZ)   /* 4096 */

#define BLK_M  128
#define BLK_N  256
#define BLK_K  32
#define LDSS   40               /* 32 + 8 bf16 pad -> 80B row stride, conflict-friendly */

union FragU { uint4 q[2]; v16bf v; };

__device__ __forceinline__ float sigmoidf_(float x) {
  return 1.0f / (1.0f + __expf(-x));
}

// Issue one 16B async global->LDS copy (gfx1250 GLOBAL_LOAD_ASYNC_TO_LDS_B128, GVS form).
// ldsAddr: LDS byte offset (low 32 bits of generic shared address); gOff: byte offset vs base.
__device__ __forceinline__ void async_b128(uint32_t ldsAddr, const void* base, uint32_t gOff) {
  asm volatile("global_load_async_to_lds_b128 %0, %1, %2"
               :: "v"(ldsAddr), "v"(gOff), "s"(base) : "memory");
}

#define WAIT_ASYNCCNT(n) asm volatile("s_wait_asynccnt " #n ::: "memory")

// ---------------- LayerNorm: fp32 x[row, C] -> bf16 xn ----------------
__global__ __launch_bounds__(256) void ln_bf16_kernel(
    const float* __restrict__ x, const float* __restrict__ gamma,
    const float* __restrict__ beta, __bf16* __restrict__ out) {
  const int row = blockIdx.x;
  const float* xr = x + (size_t)row * C_SZ;
  float s = 0.f, ss = 0.f;
  for (int i = threadIdx.x; i < C_SZ; i += 256) {
    float v = xr[i];
    s += v; ss += v * v;
  }
  __shared__ float rs[256], rq[256];
  rs[threadIdx.x] = s; rq[threadIdx.x] = ss;
  __syncthreads();
  for (int off = 128; off > 0; off >>= 1) {
    if (threadIdx.x < off) {
      rs[threadIdx.x] += rs[threadIdx.x + off];
      rq[threadIdx.x] += rq[threadIdx.x + off];
    }
    __syncthreads();
  }
  const float mean = rs[0] * (1.0f / C_SZ);
  const float var  = rq[0] * (1.0f / C_SZ) - mean * mean;
  const float inv  = rsqrtf(var + 1e-5f);
  __bf16* orow = out + (size_t)row * C_SZ;
  for (int i = threadIdx.x; i < C_SZ; i += 256) {
    float v = (xr[i] - mean) * inv * gamma[i] + beta[i];
    orow[i] = (__bf16)v;
  }
}

// ---------------- fp32 -> bf16 convert ----------------
__global__ __launch_bounds__(256) void cvt_bf16_kernel(
    const float* __restrict__ in, __bf16* __restrict__ out, int n) {
  int i = blockIdx.x * 256 + threadIdx.x;
  if (i < n) out[i] = (__bf16)in[i];
}

// ---------------- WMMA GEMM: out[M,N] = A_bf16[M,K] @ W_bf16[N,K]^T ----------------
// 256 threads = 8 waves (2x4); block tile 128x256; wave tile 64x64 (4x4 frags); BK=32.
// Double-buffered LDS staged with GLOBAL_LOAD_ASYNC_TO_LDS_B128 (ASYNCcnt-tracked prefetch).
template <bool BIAS, bool SIG, bool OUTF, bool OUTB>
__global__ __launch_bounds__(256) void gemm_wmma_kernel(
    const __bf16* __restrict__ A, const __bf16* __restrict__ W,
    const float* __restrict__ bias,
    float* __restrict__ outF, __bf16* __restrict__ outB,
    int Mtot, int Ntot, int Ktot) {
  __shared__ __align__(16) __bf16 As[2][BLK_M * LDSS];   // 2 x 10 KB
  __shared__ __align__(16) __bf16 Bs[2][BLK_N * LDSS];   // 2 x 20 KB

  const int tid   = threadIdx.x;
  const int lane  = tid & 31;
  const int wave  = tid >> 5;       // 0..7
  const int waveM = wave >> 2;      // 0..1  (64-row strips)
  const int waveN = wave & 3;       // 0..3  (64-col strips)
  const int bm    = blockIdx.x * BLK_M;
  const int bn    = blockIdx.y * BLK_N;

  const int rsel = lane & 15;
  const int hb   = lane >> 4;       // lane half

  // staging assignment: A -> 1 chunk of 32B per thread, B -> 2 chunks of 32B per thread
  const int aRow  = tid >> 1;
  const int aHalf = (tid & 1) * 16;
  const int bRow0 = tid >> 1;              // chunk 0: g = tid
  const int bHalf0= (tid & 1) * 16;
  const int bRow1 = (256 + tid) >> 1;      // chunk 1: g = 256 + tid
  const int bHalf1= ((256 + tid) & 1) * 16;

  const uint32_t asBase0 = (uint32_t)(uintptr_t)&As[0][0];
  const uint32_t asBase1 = (uint32_t)(uintptr_t)&As[1][0];
  const uint32_t bsBase0 = (uint32_t)(uintptr_t)&Bs[0][0];
  const uint32_t bsBase1 = (uint32_t)(uintptr_t)&Bs[1][0];

  const uint32_t aLds  = (uint32_t)((aRow  * LDSS + aHalf ) * sizeof(__bf16));
  const uint32_t bLds0 = (uint32_t)((bRow0 * LDSS + bHalf0) * sizeof(__bf16));
  const uint32_t bLds1 = (uint32_t)((bRow1 * LDSS + bHalf1) * sizeof(__bf16));

  auto issueTile = [&](int k0, int bf) {
    const uint32_t ab = bf ? asBase1 : asBase0;
    const uint32_t bb = bf ? bsBase1 : bsBase0;
    const uint32_t aOff  = (uint32_t)(((uint32_t)(bm + aRow ) * (uint32_t)Ktot + k0 + aHalf ) * sizeof(__bf16));
    const uint32_t bOff0 = (uint32_t)(((uint32_t)(bn + bRow0) * (uint32_t)Ktot + k0 + bHalf0) * sizeof(__bf16));
    const uint32_t bOff1 = (uint32_t)(((uint32_t)(bn + bRow1) * (uint32_t)Ktot + k0 + bHalf1) * sizeof(__bf16));
    async_b128(ab + aLds,        A, aOff);        // 6 async instructions / tile / wave
    async_b128(ab + aLds  + 16u, A, aOff  + 16u);
    async_b128(bb + bLds0,       W, bOff0);
    async_b128(bb + bLds0 + 16u, W, bOff0 + 16u);
    async_b128(bb + bLds1,       W, bOff1);
    async_b128(bb + bLds1 + 16u, W, bOff1 + 16u);
  };

  v8f acc[4][4];
  const v8f vzero = {0.f, 0.f, 0.f, 0.f, 0.f, 0.f, 0.f, 0.f};
#pragma unroll
  for (int i = 0; i < 4; ++i)
#pragma unroll
    for (int j = 0; j < 4; ++j) acc[i][j] = vzero;

  const int nk = Ktot / BLK_K;
  issueTile(0, 0);                               // prologue prefetch

  for (int it = 0; it < nk; ++it) {
    const int cur = it & 1;
    if (it + 1 < nk) {
      issueTile((it + 1) * BLK_K, cur ^ 1);      // prefetch next tile into other buffer
      WAIT_ASYNCCNT(6);                          // wait for CURRENT tile (in-order completion)
    } else {
      WAIT_ASYNCCNT(0);
    }
    __syncthreads();

    const __bf16* Ab = As[cur];
    const __bf16* Bb = Bs[cur];

    // ---- fragments (ISA K-striping: A lanes0-15 K{0-7,16-23}, lanes16-31 K{8-15,24-31};
    //      B lanes0-15 K{0-7,16-23}, lanes16-31 K{8-15,24-31}) ----
    v16bf aF[4], bF[4];
#pragma unroll
    for (int fm = 0; fm < 4; ++fm) {
      const __bf16* p = Ab + (waveM * 64 + fm * 16 + rsel) * LDSS;
      FragU u;
      u.q[0] = *(const uint4*)(p + 8 * hb);
      u.q[1] = *(const uint4*)(p + 16 + 8 * hb);
      aF[fm] = u.v;
    }
#pragma unroll
    for (int fn = 0; fn < 4; ++fn) {
      const __bf16* p = Bb + (waveN * 64 + fn * 16 + rsel) * LDSS;
      FragU u;
      u.q[0] = *(const uint4*)(p + 8 * hb);
      u.q[1] = *(const uint4*)(p + 16 + 8 * hb);
      bF[fn] = u.v;
    }

#pragma unroll
    for (int fm = 0; fm < 4; ++fm)
#pragma unroll
      for (int fn = 0; fn < 4; ++fn)
        acc[fm][fn] = __builtin_amdgcn_wmma_f32_16x16x32_bf16(
            false, aF[fm], false, bF[fn], (short)0, acc[fm][fn], false, false);
    __syncthreads();                             // buffer reuse guard
  }

  // ---- epilogue: D layout: VGPR j -> row j (+8 upper lane half), col = lane&15 ----
  const int rhi = hb * 8;
#pragma unroll
  for (int fm = 0; fm < 4; ++fm) {
#pragma unroll
    for (int fn = 0; fn < 4; ++fn) {
      const int col = bn + waveN * 64 + fn * 16 + rsel;
      const float bv = BIAS ? bias[col] : 0.f;
#pragma unroll
      for (int j = 0; j < 8; ++j) {
        const int row = bm + waveM * 64 + fm * 16 + rhi + j;
        float v = acc[fm][fn][j] + bv;
        if (SIG) v = sigmoidf_(v);
        const size_t off = (size_t)row * Ntot + col;
        if (OUTF) outF[off] = v;
        if (OUTB) outB[off] = (__bf16)v;
      }
    }
  }
}

// ---------------- sequential WKV scan: one block per (b,h), 64 threads (thread = col e) ----
__global__ __launch_bounds__(64) void wkv_scan_kernel(
    const float* __restrict__ kf, const float* __restrict__ vf,
    const float* __restrict__ wf, const float* __restrict__ rf,
    __bf16* __restrict__ outs, float* __restrict__ stateOut) {
  const int bh = blockIdx.x;
  const int b = bh / H_SZ;
  const int h = bh % H_SZ;
  const int e = threadIdx.x;                       // 0..63
  const size_t base = (size_t)b * T_SZ * C_SZ + (size_t)h * HS_SZ;

  float s[HS_SZ];
#pragma unroll
  for (int d = 0; d < HS_SZ; ++d) s[d] = 0.f;

  __shared__ float kk[HS_SZ], ww[HS_SZ], rr[HS_SZ];

  for (int t = 0; t < T_SZ; ++t) {
    const size_t off = base + (size_t)t * C_SZ;
    kk[e] = kf[off + e];
    ww[e] = wf[off + e];
    rr[e] = rf[off + e];
    const float ve = vf[off + e];
    __syncthreads();
    float acc = 0.f;
#pragma unroll
    for (int d = 0; d < HS_SZ; ++d) {
      s[d] = fmaf(1.f - ww[d], s[d], kk[d] * ve);  // row-wise decay + k⊗v
      acc  = fmaf(rr[d], s[d], acc);               // readout r · state
    }
    outs[off + e] = (__bf16)acc;
    __syncthreads();
  }
  // final state [B,H,HS,HS]
#pragma unroll
  for (int d = 0; d < HS_SZ; ++d)
    stateOut[(((size_t)b * H_SZ + h) * HS_SZ + d) * HS_SZ + e] = s[d];
}

extern "C" void kernel_launch(void* const* d_in, const int* in_sizes, int n_in,
                              void* d_out, int out_size, void* d_ws, size_t ws_size,
                              hipStream_t stream) {
  (void)in_sizes; (void)n_in; (void)out_size; (void)ws_size;
  const float* x   = (const float*)d_in[0];
  const float* gam = (const float*)d_in[1];
  const float* bet = (const float*)d_in[2];
  const float* Wsrc[6] = { (const float*)d_in[3],   // Wx
                           (const float*)d_in[4],   // Ww
                           (const float*)d_in[6],   // Wk
                           (const float*)d_in[7],   // Wv
                           (const float*)d_in[8],   // Wr
                           (const float*)d_in[9] }; // Wo
  const float* bw = (const float*)d_in[5];

  float* y        = (float*)d_out;                       // [B,T,C]
  float* stateOut = y + (size_t)M_ROWS * C_SZ;           // [B,H,HS,HS]

  // ---- workspace carve-up ----
  char* ws = (char*)d_ws;
  const size_t actEl = (size_t)M_ROWS * C_SZ;            // 8,388,608
  const size_t wEl   = (size_t)C_SZ * C_SZ;              // 4,194,304
  __bf16* xn = (__bf16*)ws;            ws += actEl * sizeof(__bf16);
  __bf16* wbf[6];
  for (int i = 0; i < 6; ++i) { wbf[i] = (__bf16*)ws; ws += wEl * sizeof(__bf16); }
  __bf16* xp   = (__bf16*)ws;          ws += actEl * sizeof(__bf16);
  float*  wdec = (float*)ws;           ws += actEl * sizeof(float);
  float*  kf   = (float*)ws;           ws += actEl * sizeof(float);
  float*  vf   = (float*)ws;           ws += actEl * sizeof(float);
  float*  rf   = (float*)ws;           ws += actEl * sizeof(float);
  __bf16* outs = (__bf16*)ws;          ws += actEl * sizeof(__bf16);

  // 1) LayerNorm -> bf16
  ln_bf16_kernel<<<M_ROWS, 256, 0, stream>>>(x, gam, bet, xn);

  // 2) weights -> bf16
  const int cvtGrid = (int)((wEl + 255) / 256);
  for (int i = 0; i < 6; ++i)
    cvt_bf16_kernel<<<cvtGrid, 256, 0, stream>>>(Wsrc[i], wbf[i], (int)wEl);

  const dim3 gGrid(M_ROWS / BLK_M, C_SZ / BLK_N);        // 32 x 8

  // 3) xp = xn @ Wx^T (bf16 out)
  gemm_wmma_kernel<false, false, false, true><<<gGrid, 256, 0, stream>>>(
      xn, wbf[0], nullptr, nullptr, xp, M_ROWS, C_SZ, C_SZ);
  // 4) w_dec = sigmoid(xp @ Ww^T + bw) (f32 out)
  gemm_wmma_kernel<true, true, true, false><<<gGrid, 256, 0, stream>>>(
      xp, wbf[1], bw, wdec, nullptr, M_ROWS, C_SZ, C_SZ);
  // 5) k = xn @ Wk^T ; v = xn @ Wv^T (f32)
  gemm_wmma_kernel<false, false, true, false><<<gGrid, 256, 0, stream>>>(
      xn, wbf[2], nullptr, kf, nullptr, M_ROWS, C_SZ, C_SZ);
  gemm_wmma_kernel<false, false, true, false><<<gGrid, 256, 0, stream>>>(
      xn, wbf[3], nullptr, vf, nullptr, M_ROWS, C_SZ, C_SZ);
  // 6) r = sigmoid(xn @ Wr^T) (f32)
  gemm_wmma_kernel<false, true, true, false><<<gGrid, 256, 0, stream>>>(
      xn, wbf[4], nullptr, rf, nullptr, M_ROWS, C_SZ, C_SZ);

  // 7) sequential scan -> outs (bf16) + final state (f32 tail of d_out)
  wkv_scan_kernel<<<B_SZ * H_SZ, HS_SZ, 0, stream>>>(kf, vf, wdec, rf, outs, stateOut);

  // 8) y = outs @ Wo^T -> d_out (f32)
  gemm_wmma_kernel<false, false, true, false><<<gGrid, 256, 0, stream>>>(
      outs, wbf[5], nullptr, y, nullptr, M_ROWS, C_SZ, C_SZ);
}
